// GraphConvolution_12446815224390
// MI455X (gfx1250) — matrix operand order repaired and checked by reading, not compiled, not run on hardware
//
#include <hip/hip_runtime.h>
#include <stdint.h>

typedef __attribute__((ext_vector_type(16))) __bf16 v16bf;
typedef __attribute__((ext_vector_type(8)))  float  v8f;

#define IN_F   512
#define OUT_F  256
// packed-W plane size (elements): 16 ktiles * 16 ntiles * 32 lanes * 16 elems
#define WP_PLANE (16 * 16 * 32 * 16)

// ---------------------------------------------------------------------------
// Kernel 0: pack W (fp32 [512,256]) into bf16 hi/lo planes in the exact
// V_WMMA 32x16 B-fragment lane layout: lane group (lane>>4) selects K half,
// lane&15 selects column, 16 contiguous bf16 per lane.
// ---------------------------------------------------------------------------
__global__ void wpack_kernel(const float* __restrict__ W, __bf16* __restrict__ wp) {
    int tid = blockIdx.x * blockDim.x + threadIdx.x;      // (kt*16 + nt)*32 + lane
    if (tid >= 16 * 16 * 32) return;
    int lane  = tid & 31;
    int kt    = tid >> 9;                                  // 0..15
    int col   = ((tid >> 5) & 15) * 16 + (lane & 15);      // ntile*16 + n
    int kbase = kt * 32 + (lane >> 4) * 8;                 // K octet per lane group (+16 for elems 8..15)
    __bf16* hi = wp + (size_t)tid * 16;
    __bf16* lo = hi + WP_PLANE;
#pragma unroll
    for (int e = 0; e < 16; ++e) {
        float f  = W[(size_t)(kbase + (e & 7) + (e >> 3) * 16) * OUT_F + col];
        __bf16 h = (__bf16)f;
        hi[e] = h;
        lo[e] = (__bf16)(f - (float)h);
    }
}

// ---------------------------------------------------------------------------
// Kernel 1: out[i] = bias[i % 256]  (fused bias + output init)
// ---------------------------------------------------------------------------
__global__ void out_init_kernel(float* __restrict__ out, const float* __restrict__ bias,
                                size_t total) {
    size_t i = (size_t)blockIdx.x * blockDim.x + threadIdx.x;
    if (i < total) out[i] = bias[i & (OUT_F - 1)];
}

// ---------------------------------------------------------------------------
// Kernel 2: support = x @ W via bf16x3 split WMMA (near-fp32 accuracy).
// Block = 256 thr = 8 waves laid out 2(M) x 4(N); wave tile = 16M x 64N.
// ---------------------------------------------------------------------------
__global__ void __launch_bounds__(256) gemm_kernel(const float* __restrict__ x,
                                                   const __bf16* __restrict__ wp,
                                                   float* __restrict__ support,
                                                   int nrows) {
    const int lane  = threadIdx.x & 31;
    const int wave  = threadIdx.x >> 5;
    const int waveM = wave & 1;
    const int waveN = wave >> 1;                 // 0..3
    const int rowTile = blockIdx.x * 32 + waveM * 16;
    if (rowTile >= nrows) return;                // wave-uniform: EXEC stays all-1s
    const int kg   = lane >> 4;                  // A lane group (K half)
    const int rowA = rowTile + (lane & 15);
    const float* xr = x + (size_t)rowA * IN_F;

    v8f acc[4] = {};

    for (int kt = 0; kt < IN_F / 32; ++kt) {
        // --- A fragment: ISA 16-bit A 16x32 layout ---
        // elems 0..7  <- K = kt*32 + kg*8 + (0..7)
        // elems 8..15 <- K = kt*32 + 16 + kg*8 + (0..7)
        const int k0 = kt * 32 + kg * 8;
        float4 f0 = *(const float4*)(xr + k0);
        float4 f1 = *(const float4*)(xr + k0 + 4);
        float4 f2 = *(const float4*)(xr + k0 + 16);
        float4 f3 = *(const float4*)(xr + k0 + 20);
        float fa[16] = { f0.x, f0.y, f0.z, f0.w, f1.x, f1.y, f1.z, f1.w,
                         f2.x, f2.y, f2.z, f2.w, f3.x, f3.y, f3.z, f3.w };
        v16bf ah, al;
#pragma unroll
        for (int e = 0; e < 16; ++e) {
            __bf16 h = (__bf16)fa[e];
            ah[e] = h;
            al[e] = (__bf16)(fa[e] - (float)h);
        }
#pragma unroll
        for (int i = 0; i < 4; ++i) {
            const int nt = waveN * 4 + i;
            const __bf16* bp = wp + ((size_t)(kt * 16 + nt) * 32 + lane) * 16;
            v16bf bh = *(const v16bf*)bp;
            v16bf bl = *(const v16bf*)(bp + WP_PLANE);
            // bf16x3: hi*hi + lo*hi + hi*lo, fp32 accumulate
            acc[i] = __builtin_amdgcn_wmma_f32_16x16x32_bf16(false, ah, false, bh,
                                                             (short)0, acc[i], false, false);
            acc[i] = __builtin_amdgcn_wmma_f32_16x16x32_bf16(false, al, false, bh,
                                                             (short)0, acc[i], false, false);
            acc[i] = __builtin_amdgcn_wmma_f32_16x16x32_bf16(false, ah, false, bl,
                                                             (short)0, acc[i], false, false);
        }
    }

    // C/D layout: VGPR r -> M = r + 8*(lane>=16), N = lane&15
    const int half = lane >> 4;
    const int nc   = lane & 15;
#pragma unroll
    for (int i = 0; i < 4; ++i) {
        const int col = (waveN * 4 + i) * 16 + nc;
#pragma unroll
        for (int r = 0; r < 8; ++r) {
            support[(size_t)(rowTile + half * 8 + r) * OUT_F + col] = acc[i][r];
        }
    }
}

// ---------------------------------------------------------------------------
// Kernel 3: SpMM — one wave per edge; lane owns 8 consecutive features.
// Coalesced 1KB gather from support (L2-resident), then native fp32 atomics.
// ---------------------------------------------------------------------------
__global__ void __launch_bounds__(256) spmm_kernel(const float* __restrict__ support,
                                                   const int* __restrict__ erow,
                                                   const int* __restrict__ ecol,
                                                   const float* __restrict__ ew,
                                                   float* __restrict__ out,
                                                   int nedges) {
    const int lane = threadIdx.x & 31;
    const int e = (int)(((size_t)blockIdx.x * blockDim.x + threadIdx.x) >> 5);
    if (e >= nedges) return;
    const int   r = erow[e];
    const int   c = ecol[e];
    const float w = ew[e];
    const float4* sp = (const float4*)(support + (size_t)c * OUT_F + lane * 8);
    float4 a = sp[0], b = sp[1];
    float vals[8] = { a.x * w, a.y * w, a.z * w, a.w * w,
                      b.x * w, b.y * w, b.z * w, b.w * w };
    float* op = out + (size_t)r * OUT_F + lane * 8;
#pragma unroll
    for (int j = 0; j < 8; ++j) {
        asm volatile("global_atomic_add_f32 %0, %1, off"
                     :: "v"(op + j), "v"(vals[j]) : "memory");
    }
}

// ---------------------------------------------------------------------------
extern "C" void kernel_launch(void* const* d_in, const int* in_sizes, int n_in,
                              void* d_out, int out_size, void* d_ws, size_t ws_size,
                              hipStream_t stream) {
    const float* x    = (const float*)d_in[0];
    const float* W    = (const float*)d_in[1];
    const float* bias = (const float*)d_in[2];
    const int*   erow = (const int*)d_in[3];
    const int*   ecol = (const int*)d_in[4];
    const float* ew   = (const float*)d_in[5];
    float* out = (float*)d_out;

    const int nrows  = in_sizes[0] / IN_F;   // 100000
    const int nedges = in_sizes[3];          // 3200000

    // workspace layout: [ support : nrows*256 fp32 ][ packed W : 2 planes bf16 ]
    float*  support = (float*)d_ws;
    __bf16* wp = (__bf16*)((char*)d_ws + (size_t)nrows * OUT_F * sizeof(float));

    wpack_kernel<<<(16 * 16 * 32 + 255) / 256, 256, 0, stream>>>(W, wp);

    size_t total = (size_t)nrows * OUT_F;
    out_init_kernel<<<(unsigned)((total + 255) / 256), 256, 0, stream>>>(out, bias, total);

    gemm_kernel<<<(nrows + 31) / 32, 256, 0, stream>>>(x, wp, support, nrows);

    unsigned spmm_blocks = (unsigned)(((size_t)nedges + 7) / 8);   // 8 edges/block
    spmm_kernel<<<spmm_blocks, 256, 0, stream>>>(support, erow, ecol, ew, out, nedges);
}
